// NonLocalBlock_63024350102345
// MI455X (gfx1250) — matrix operand order repaired and checked by reading, not compiled, not run on hardware
//
#include <hip/hip_runtime.h>

// ---------------------------------------------------------------------------
// NonLocalBlock: x[4,256,64,64] fp32, four 1x1 convs + softmax attention.
// All GEMMs run on v_wmma_f32_16x16x32_bf16 (fp32 accumulate).
// Attention is fused flash-style with double-buffered LDS staging; K chunks
// use CDNA5 async global->LDS copies (ASYNCcnt).
// ---------------------------------------------------------------------------

typedef __attribute__((ext_vector_type(16))) __bf16 v16bf;
typedef __attribute__((ext_vector_type(8)))  __bf16 v8bf;
typedef __attribute__((ext_vector_type(8)))  float  v8f;
typedef __attribute__((ext_vector_type(4)))  float  v4f;
typedef __attribute__((ext_vector_type(4)))  int    v4i;

#define WMMA_BF16(A, B, C) \
  __builtin_amdgcn_wmma_f32_16x16x32_bf16(false, (A), false, (B), (short)0, (C), false, false)

#define NCH   256          // channels
#define NSP   4096         // 64*64 spatial
#define PERB  (NCH * NSP)  // elements per batch per tensor

// ---- CDNA5 async global->LDS copy (tracked by ASYNCcnt), with safe fallback
#if __has_builtin(__builtin_amdgcn_global_load_async_to_lds_b128)
#define HAVE_ASYNC_LDS 1
#endif

__device__ __forceinline__ void copy16_g2l(__bf16* lds, const __bf16* g) {
#ifdef HAVE_ASYNC_LDS
  // signature (from clang diagnostic): (AS1 v4i*, AS3 v4i*, imm offset, cpol)
  void* gp = (void*)g;  // drop const, then addrspace-cast
  __builtin_amdgcn_global_load_async_to_lds_b128(
      (__attribute__((address_space(1))) v4i*)gp,
      (__attribute__((address_space(3))) v4i*)lds, 0, 0);
#else
  *(v8bf*)lds = *(const v8bf*)g;
#endif
}

__device__ __forceinline__ void wait_async_lds() {
#ifdef HAVE_ASYNC_LDS
#if __has_builtin(__builtin_amdgcn_s_wait_asynccnt)
  __builtin_amdgcn_s_wait_asynccnt(0);
#else
  asm volatile("s_wait_asynccnt 0x0" ::: "memory");
#endif
#endif
}

// ---------------------------------------------------------------------------
// Kernel 1: fused Q/K/V projection.  Out[o,s] = sum_c W[o,c]*x[b,c,s] + bias[o]
// One wave -> one 16x16 output tile, K-loop of 8 bf16 WMMAs.
// Output stored bf16 in NCHW layout; flat view [n,c] feeds attention directly.
// ---------------------------------------------------------------------------
__global__ __launch_bounds__(256)
void qkv_proj_kernel(const float* __restrict__ x,
                     const float* __restrict__ Wq, const float* __restrict__ bq,
                     const float* __restrict__ Wk, const float* __restrict__ bk,
                     const float* __restrict__ Wv, const float* __restrict__ bv,
                     __bf16* __restrict__ Qb, __bf16* __restrict__ Kb,
                     __bf16* __restrict__ Vb)
{
  const int lane = threadIdx.x & 31;
  const int wid  = blockIdx.x * 8 + (threadIdx.x >> 5);
  const int st = wid & 255;          // spatial tile (16 cols)
  const int mt = (wid >> 8) & 15;    // out-channel tile (16 rows)
  const int b  = (wid >> 12) & 3;    // batch
  const int p  = wid >> 14;          // 0=Q 1=K 2=V
  const int l16 = lane & 15;
  const int hi  = lane >> 4;
  const int s0 = st * 16;
  const int m0 = mt * 16;

  const float* W    = (p == 0) ? Wq : ((p == 1) ? Wk : Wv);
  const float* bias = (p == 0) ? bq : ((p == 1) ? bk : bv);
  __bf16* out = ((p == 0) ? Qb : ((p == 1) ? Kb : Vb)) + (size_t)b * PERB;
  const float* xb = x + (size_t)b * PERB;

  v8f acc = {};
  for (int k0 = 0; k0 < NCH; k0 += 32) {
    // A fragment (16x32 of W): lane<16 holds K{k0..k0+15}, hi lanes +16
    const float* wr = W + (m0 + l16) * NCH + k0 + hi * 16;
    v4f w0 = *(const v4f*)(wr);
    v4f w1 = *(const v4f*)(wr + 4);
    v4f w2 = *(const v4f*)(wr + 8);
    v4f w3 = *(const v4f*)(wr + 12);
    v16bf a;
#pragma unroll
    for (int j = 0; j < 4; ++j) {
      a[j]      = (__bf16)w0[j];
      a[4 + j]  = (__bf16)w1[j];
      a[8 + j]  = (__bf16)w2[j];
      a[12 + j] = (__bf16)w3[j];
    }
    // B fragment (32x16 of x): lane = column s0+l16; K = hi*16 + (0..15)
    v16bf bb;
    const float* xc = xb + (size_t)(k0 + hi * 16) * NSP + s0 + l16;
#pragma unroll
    for (int j = 0; j < 16; ++j) bb[j] = (__bf16)xc[(size_t)j * NSP];
    acc = WMMA_BF16(a, bb, acc);
  }
#pragma unroll
  for (int i = 0; i < 8; ++i) {
    const int r = m0 + i + hi * 8;            // C/D layout: vgpr i -> row i (+8 hi)
    const float v = acc[i] + bias[r];
    out[(size_t)r * NSP + s0 + l16] = (__bf16)v;
  }
}

// ---------------------------------------------------------------------------
// Kernel 2: fused flash attention per batch.
// Block = 8 waves; wave owns 16 query rows (Q kept as 8 A-fragments in VGPRs).
// Double-buffered LDS staging: K via async global->LDS (no VGPR round trip),
// V transposed through registers (loads issued before compute, ds stores after)
// so both copies overlap the 32 WMMAs of the current chunk.
// ---------------------------------------------------------------------------
__global__ __launch_bounds__(256)
void attn_kernel(const __bf16* __restrict__ Qb, const __bf16* __restrict__ Kb,
                 const __bf16* __restrict__ Vb, __bf16* __restrict__ Ab)
{
  __shared__ __align__(16) __bf16 Klds[2][32 * 256];   // [m_local][c]
  __shared__ __align__(16) __bf16 Vlds[2][256 * 32];   // [c][m_local] (transposed)
  __shared__ __align__(16) __bf16 Plds[8][16 * 32];    // per-wave P scratch

  const int tid  = threadIdx.x;
  const int lane = tid & 31;
  const int w    = tid >> 5;
  const int l16  = lane & 15;
  const int hi   = lane >> 4;
  const int b    = blockIdx.x >> 5;                    // 32 blocks per batch
  const int nblk = blockIdx.x & 31;
  const int n0   = nblk * 128 + w * 16;                // this wave's query rows
  const size_t bbase = (size_t)b * PERB;

  // cooperative staging positions
  const int vm  = tid >> 3;          // V source row (m_local) 0..31
  const int vc0 = (tid & 7) * 32;    // V source channel block 0..224

  // Q A-fragments for all 8 K-steps (rows contiguous 256-bf16 in the flat view)
  v16bf qf[8];
#pragma unroll
  for (int kt = 0; kt < 8; ++kt) {
    const __bf16* qp = Qb + bbase + (size_t)(n0 + l16) * NCH + kt * 32 + hi * 16;
    v8bf q0 = *(const v8bf*)qp;
    v8bf q1 = *(const v8bf*)(qp + 8);
#pragma unroll
    for (int j = 0; j < 8; ++j) { qf[kt][j] = q0[j]; qf[kt][8 + j] = q1[j]; }
  }

  float m_run[8], l_run[8];
#pragma unroll
  for (int i = 0; i < 8; ++i) { m_run[i] = -3.0e38f; l_run[i] = 0.0f; }
  v8f O[16];
#pragma unroll
  for (int t = 0; t < 16; ++t) O[t] = (v8f){};

  v8bf vreg[4];

  auto stage_k = [&](int mc, int buf) {      // async: LDS written by DMA
    const __bf16* src = Kb + bbase + (size_t)mc * NCH + tid * 32;
    __bf16* dst = &Klds[buf][tid * 32];
#pragma unroll
    for (int j = 0; j < 4; ++j) copy16_g2l(dst + j * 8, src + j * 8);
  };
  auto load_v = [&](int mc) {                // issue global loads into registers
    const __bf16* src = Vb + bbase + (size_t)(mc + vm) * NCH + vc0;
#pragma unroll
    for (int j = 0; j < 4; ++j) vreg[j] = *(const v8bf*)(src + j * 8);
  };
  auto store_v = [&](int buf) {              // transposed scatter into LDS
#pragma unroll
    for (int j = 0; j < 4; ++j)
#pragma unroll
      for (int e = 0; e < 8; ++e)
        Vlds[buf][(vc0 + j * 8 + e) * 32 + vm] = vreg[j][e];
  };

  // prologue: fill buffer 0 with chunk 0
  stage_k(0, 0);
  load_v(0);
  store_v(0);
  wait_async_lds();
  __syncthreads();

  for (int mc = 0, it = 0; mc < NSP; mc += 32, ++it) {
    const int cur = it & 1;
    const int nxt = cur ^ 1;
    const bool more = (mc + 32) < NSP;
    if (more) { stage_k(mc + 32, nxt); load_v(mc + 32); }  // overlaps compute

    // ---- S = Q . K^T for two 16x16 key tiles (16 WMMAs) ----
    const __bf16* Kc = Klds[cur];
    v8f S0 = {}, S1 = {};
#pragma unroll
    for (int kt = 0; kt < 8; ++kt) {
      // B layout: lane = column (key m_local), K = hi*16 + (0..15) contiguous
      const __bf16* k0p = Kc + l16 * NCH + kt * 32 + hi * 16;
      v8bf a0 = *(const v8bf*)k0p;
      v8bf a1 = *(const v8bf*)(k0p + 8);
      v16bf bk0;
#pragma unroll
      for (int j = 0; j < 8; ++j) { bk0[j] = a0[j]; bk0[8 + j] = a1[j]; }
      S0 = WMMA_BF16(qf[kt], bk0, S0);

      const __bf16* k1p = Kc + (16 + l16) * NCH + kt * 32 + hi * 16;
      v8bf c0v = *(const v8bf*)k1p;
      v8bf c1v = *(const v8bf*)(k1p + 8);
      v16bf bk1;
#pragma unroll
      for (int j = 0; j < 8; ++j) { bk1[j] = c0v[j]; bk1[8 + j] = c1v[j]; }
      S1 = WMMA_BF16(qf[kt], bk1, S1);
    }

    // ---- online softmax over this 32-key chunk ----
    float alpha[8];
#pragma unroll
    for (int i = 0; i < 8; ++i) {
      float r = fmaxf(S0[i], S1[i]);
      for (int off = 1; off < 16; off <<= 1)
        r = fmaxf(r, __shfl_xor(r, off, 16));     // row lives in a 16-lane group
      const float mn = fmaxf(m_run[i], r);
      alpha[i] = __expf(m_run[i] - mn);
      const float p0 = __expf(S0[i] - mn);
      const float p1 = __expf(S1[i] - mn);
      S0[i] = p0; S1[i] = p1;
      float rs = p0 + p1;
      for (int off = 1; off < 16; off <<= 1)
        rs += __shfl_xor(rs, off, 16);
      l_run[i] = l_run[i] * alpha[i] + rs;
      m_run[i] = mn;
    }
#pragma unroll
    for (int t = 0; t < 16; ++t)
#pragma unroll
      for (int i = 0; i < 8; ++i) O[t][i] *= alpha[i];

    // ---- transpose P (C/D layout -> A layout) through wave-private LDS ----
    __bf16* Ps = Plds[w];
#pragma unroll
    for (int i = 0; i < 8; ++i) {
      const int r = i + hi * 8;
      Ps[r * 32 + l16]      = (__bf16)S0[i];
      Ps[r * 32 + 16 + l16] = (__bf16)S1[i];
    }
    v16bf pf;   // same-wave LDS ops complete in order (DScnt in-order)
    {
      const __bf16* pp = Ps + l16 * 32 + hi * 16;
      v8bf p0 = *(const v8bf*)pp;
      v8bf p1 = *(const v8bf*)(pp + 8);
#pragma unroll
      for (int j = 0; j < 8; ++j) { pf[j] = p0[j]; pf[8 + j] = p1[j]; }
    }

    // ---- O += P . V   (16 WMMAs; B-frags contiguous from transposed Vlds) ----
    const __bf16* Vc = Vlds[cur];
#pragma unroll
    for (int t = 0; t < 16; ++t) {
      const __bf16* vp = Vc + (t * 16 + l16) * 32 + hi * 16;
      v8bf v0 = *(const v8bf*)vp;
      v8bf v1 = *(const v8bf*)(vp + 8);
      v16bf bv;
#pragma unroll
      for (int j = 0; j < 8; ++j) { bv[j] = v0[j]; bv[8 + j] = v1[j]; }
      O[t] = WMMA_BF16(pf, bv, O[t]);
    }

    if (more) store_v(nxt);   // V global loads were in flight during compute
    wait_async_lds();         // K async copies for chunk mc+32 complete
    __syncthreads();          // (also orders the V ds stores for all waves)
  }

  // ---- normalize + store attn rows (flat [n,c] view, bf16) ----
  float inv[8];
#pragma unroll
  for (int i = 0; i < 8; ++i) inv[i] = 1.0f / l_run[i];
#pragma unroll
  for (int t = 0; t < 16; ++t)
#pragma unroll
    for (int i = 0; i < 8; ++i)
      Ab[bbase + (size_t)(n0 + i + hi * 8) * NCH + t * 16 + l16] =
          (__bf16)(O[t][i] * inv[i]);
}

// ---------------------------------------------------------------------------
// Kernel 3: output projection + residual.
// attn raw-viewed back to NCHW: A_nchw[c][s] = Ab_flat[c*4096 + s].
// out[b,o,s] = x[b,o,s] + bu[o] + sum_c Wu[o,c] * A_nchw[b,c,s]   (fp32 out)
// ---------------------------------------------------------------------------
__global__ __launch_bounds__(256)
void out_proj_kernel(const __bf16* __restrict__ Ab, const float* __restrict__ x,
                     const float* __restrict__ Wu, const float* __restrict__ bu,
                     float* __restrict__ out)
{
  const int lane = threadIdx.x & 31;
  const int wid  = blockIdx.x * 8 + (threadIdx.x >> 5);
  const int st = wid & 255;
  const int mt = (wid >> 8) & 15;
  const int b  = wid >> 12;
  const int l16 = lane & 15;
  const int hi  = lane >> 4;
  const int s0 = st * 16;
  const int m0 = mt * 16;
  const size_t bbase = (size_t)b * PERB;

  v8f acc = {};
  for (int k0 = 0; k0 < NCH; k0 += 32) {
    const float* wr = Wu + (m0 + l16) * NCH + k0 + hi * 16;
    v4f w0 = *(const v4f*)(wr);
    v4f w1 = *(const v4f*)(wr + 4);
    v4f w2 = *(const v4f*)(wr + 8);
    v4f w3 = *(const v4f*)(wr + 12);
    v16bf a;
#pragma unroll
    for (int j = 0; j < 4; ++j) {
      a[j]      = (__bf16)w0[j];
      a[4 + j]  = (__bf16)w1[j];
      a[8 + j]  = (__bf16)w2[j];
      a[12 + j] = (__bf16)w3[j];
    }
    v16bf bb;
    const __bf16* ac = Ab + bbase + (size_t)(k0 + hi * 16) * NSP + s0 + l16;
#pragma unroll
    for (int j = 0; j < 16; ++j) bb[j] = ac[(size_t)j * NSP];
    acc = WMMA_BF16(a, bb, acc);
  }
#pragma unroll
  for (int i = 0; i < 8; ++i) {
    const int r = m0 + i + hi * 8;
    const size_t idx = bbase + (size_t)r * NSP + s0 + l16;
    out[idx] = acc[i] + bu[r] + x[idx];
  }
}

// ---------------------------------------------------------------------------
extern "C" void kernel_launch(void* const* d_in, const int* in_sizes, int n_in,
                              void* d_out, int out_size, void* d_ws, size_t ws_size,
                              hipStream_t stream)
{
  (void)in_sizes; (void)n_in; (void)out_size; (void)ws_size;
  const float* x  = (const float*)d_in[0];
  const float* Wq = (const float*)d_in[1];
  const float* bq = (const float*)d_in[2];
  const float* Wk = (const float*)d_in[3];
  const float* bk = (const float*)d_in[4];
  const float* Wv = (const float*)d_in[5];
  const float* bv = (const float*)d_in[6];
  const float* Wu = (const float*)d_in[7];
  const float* bu = (const float*)d_in[8];
  float* out = (float*)d_out;

  // workspace: Q | K | V | attn, each 4 batches * 1M bf16 = 8 MB (32 MB total)
  __bf16* Qb = (__bf16*)d_ws;
  __bf16* Kb = Qb + (size_t)4 * PERB;
  __bf16* Vb = Kb + (size_t)4 * PERB;
  __bf16* Ab = Vb + (size_t)4 * PERB;

  // 3 projections * 4 batches * 16 * 256 tiles = 49152 waves / 8 per block
  qkv_proj_kernel<<<6144, 256, 0, stream>>>(x, Wq, bq, Wk, bk, Wv, bv, Qb, Kb, Vb);
  // 4 batches * 32 blocks (8 waves * 16 query rows each = 128 rows/block)
  attn_kernel<<<128, 256, 0, stream>>>(Qb, Kb, Vb, Ab);
  // 4 batches * 4096 tiles = 16384 waves / 8 per block
  out_proj_kernel<<<2048, 256, 0, stream>>>(Ab, x, Wu, bu, out);
}
